// TensorRF_7310034338516
// MI455X (gfx1250) — matrix operand order repaired and checked by reading, not compiled, not run on hardware
//
#include <hip/hip_runtime.h>
#include <hip/hip_bf16.h>

// ---- CDNA5 WMMA types ----
typedef __attribute__((ext_vector_type(16))) _Float16     v16h;
typedef __attribute__((ext_vector_type(8)))  float        v8f;
typedef __attribute__((ext_vector_type(4)))  unsigned int v4u;

#define RES   128
#define RANK  8
#define PE_L  6
#define ENC   256

// packed-weight layout offsets (in halfs) inside d_ws
#define SZ_L0   (16 * 2 * 512)          // 16 n-tiles, 2 k-tiles (K=47 pad 64)
#define SZ_LH   (16 * 8 * 512)          // 16 n-tiles, 8 k-tiles (K=256)
#define SZ_HEAD (1 * 8 * 512)           // 1 n-tile (sigma + rgb packed), 8 k-tiles
#define OFF_L0  0
#define OFF_L1  (SZ_L0)
#define OFF_L2  (SZ_L0 + SZ_LH)
#define OFF_L3  (SZ_L0 + 2 * SZ_LH)
#define OFF_HD  (SZ_L0 + 3 * SZ_LH)

union Frag {
    v4u  u[2];
    v16h v;
};

// ---------------------------------------------------------------------------
// Weight packing: fp32 (K x Nout, row-major) -> f16 B-fragment order.
// B-fragment layout (V_WMMA_F32_16X16X32_F16, wave32):
//   lane: column n = lane & 15 ; k = (h>=8 ? 16 : 0) + (lane>=16 ? 8 : 0) + (h & 7), h = 0..15
// Packed address: (((nt*KT + kt)*32 + lane)*16 + h)
// ---------------------------------------------------------------------------
__global__ void pack_w(const float* __restrict__ W, _Float16* __restrict__ dst,
                       int Ksrc, int KT, int Nout)
{
    int total = (Nout / 16) * KT * 512;
    for (int e = blockIdx.x * blockDim.x + threadIdx.x; e < total;
         e += gridDim.x * blockDim.x) {
        int h    = e & 15;
        int lane = (e >> 4) & 31;
        int t    = e >> 9;
        int kt   = t % KT;
        int nt   = t / KT;
        int k    = kt * 32 + ((h >> 3) << 4) + ((lane >> 4) << 3) + (h & 7);
        int n    = nt * 16 + (lane & 15);
        float v  = (k < Ksrc) ? W[k * Nout + n] : 0.0f;
        dst[e] = (_Float16)v;
    }
}

// Head: pack W_sig (256x1) and W_rgb (256x3) into one 256x16 tile:
// col 0 = sigma, cols 1..3 = rgb, cols 4..15 = 0.
__global__ void pack_head(const float* __restrict__ Wsig,
                          const float* __restrict__ Wrgb,
                          _Float16* __restrict__ dst)
{
    int e = blockIdx.x * blockDim.x + threadIdx.x;
    if (e >= SZ_HEAD) return;
    int h    = e & 15;
    int lane = (e >> 4) & 31;
    int kt   = e >> 9;
    int k    = kt * 32 + ((h >> 3) << 4) + ((lane >> 4) << 3) + (h & 7);
    int n    = lane & 15;
    float v  = 0.0f;
    if (n == 0)      v = Wsig[k];
    else if (n <= 3) v = Wrgb[k * 3 + (n - 1)];
    dst[e] = (_Float16)v;
}

// ---------------------------------------------------------------------------
// Async global->LDS weight staging (CDNA5 GLOBAL_LOAD_ASYNC_TO_LDS_B128,
// GVS mode: SGPR base + per-lane VGPR byte offset; VDST = per-lane LDS
// byte address). Tracked by ASYNCcnt. Low 32 bits of a generic pointer to
// LDS are the LDS byte offset (flat-addressing truncation rule).
// ---------------------------------------------------------------------------
__device__ inline void stage_async(_Float16* dst, const _Float16* src,
                                   int nhalfs, int tid)
{
    const void* sbase = (const void*)src;
    int n16 = nhalfs >> 3;                     // 16-byte chunks
    for (int i = tid; i < n16; i += 256) {
        unsigned int lds32 = (unsigned int)(size_t)(dst + i * 8);
        unsigned int voff  = (unsigned int)(i * 16);
        asm volatile("global_load_async_to_lds_b128 %0, %1, %2"
                     :
                     : "v"(lds32), "v"(voff), "s"(sbase)
                     : "memory");
    }
}

__device__ inline void wait_async_all()
{
    asm volatile("s_wait_asynccnt 0x0" ::: "memory");
}

// ---------------------------------------------------------------------------
// One MLP layer for one wave (16 points):
//   h_out(16x256) = relu( h_in(16 x 32*KT) * W + b )
// Two n-tiles in flight -> adjacent v_wmma are independent, hiding the
// WMMA->WMMA RAW hazard (1 NOP + 4 co-exec slots for f16).
// ---------------------------------------------------------------------------
template <int KT>
__device__ inline void mlp_layer(_Float16* hb, const _Float16* wb,
                                 const float* __restrict__ bias, int lane)
{
    const int ln15 = lane & 15;
    const int klo  = (lane < 16) ? 0 : 8;
    const int mhi  = (lane < 16) ? 0 : 8;

    v16h a[KT];
#pragma unroll
    for (int kt = 0; kt < KT; ++kt) {
        const v4u* p = (const v4u*)(hb + ln15 * ENC + kt * 32 + klo);
        Frag f; f.u[0] = p[0]; f.u[1] = p[2];
        a[kt] = f.v;
    }

    for (int nt = 0; nt < 16; nt += 2) {
        float b0 = bias[nt * 16 + ln15];
        float b1 = bias[(nt + 1) * 16 + ln15];
        v8f c0, c1;
#pragma unroll
        for (int r = 0; r < 8; ++r) { c0[r] = b0; c1[r] = b1; }

#pragma unroll
        for (int kt = 0; kt < KT; ++kt) {
            const v4u* q0 = (const v4u*)(wb + ((nt * KT + kt) * 32 + lane) * 16);
            const v4u* q1 = (const v4u*)(wb + (((nt + 1) * KT + kt) * 32 + lane) * 16);
            Frag g0; g0.u[0] = q0[0]; g0.u[1] = q0[1];
            Frag g1; g1.u[0] = q1[0]; g1.u[1] = q1[1];
            c0 = __builtin_amdgcn_wmma_f32_16x16x32_f16(
                     false, a[kt], false, g0.v, (short)0, c0, false, false);
            c1 = __builtin_amdgcn_wmma_f32_16x16x32_f16(
                     false, a[kt], false, g1.v, (short)0, c1, false, false);
        }

#pragma unroll
        for (int r = 0; r < 8; ++r) {
            int mm = r + mhi;
            float v0 = c0[r]; v0 = v0 > 0.0f ? v0 : 0.0f;
            float v1 = c1[r]; v1 = v1 > 0.0f ? v1 : 0.0f;
            hb[mm * ENC + nt * 16 + ln15]        = (_Float16)v0;
            hb[mm * ENC + (nt + 1) * 16 + ln15]  = (_Float16)v1;
        }
    }
}

// ---------------------------------------------------------------------------
// Main fused kernel: 256 threads = 8 waves; each wave owns 16 points.
// LDS: hb = 8 * 16*256 f16 activation tiles (64 KB)
//      wb = current layer packed weights (<=128 KB), async-staged
// ---------------------------------------------------------------------------
__global__ __launch_bounds__(256)
void tensorf_mlp(const float* __restrict__ X,
                 const float* __restrict__ Fxy,
                 const float* __restrict__ Fxz,
                 const float* __restrict__ Fyz,
                 const _Float16* __restrict__ Wpk,
                 const float* __restrict__ B0, const float* __restrict__ B1,
                 const float* __restrict__ B2, const float* __restrict__ B3,
                 const float* __restrict__ Bsig, const float* __restrict__ Brgb,
                 float* __restrict__ Out, int nPoints)
{
    extern __shared__ char smem[];
    _Float16* hb_all = (_Float16*)smem;                        // 64 KB
    _Float16* wb     = (_Float16*)(smem + 8 * 16 * ENC * 2);   // up to 128 KB

    const int tid  = threadIdx.x;
    const int wave = tid >> 5;
    const int lane = tid & 31;
    _Float16* hb = hb_all + wave * 16 * ENC;
    const int tileBase = blockIdx.x * 128 + wave * 16;

    // kick off layer-0 weight staging (async, no VGPR traffic) ...
    stage_async(wb, Wpk + OFF_L0, SZ_L0, tid);

    // ... and overlap it with per-point input feature construction
    if (lane < 16) {
        int idx  = tileBase + lane;
        float x0 = X[idx * 3 + 0];
        float x1 = X[idx * 3 + 1];
        float x2 = X[idx * 3 + 2];
        int ix = min(max((int)((x0 + 1.0f) * 0.5f * (RES - 1)), 0), RES - 1);
        int iy = min(max((int)((x1 + 1.0f) * 0.5f * (RES - 1)), 0), RES - 1);
        int iz = min(max((int)((x2 + 1.0f) * 0.5f * (RES - 1)), 0), RES - 1);
        _Float16* row = hb + lane * ENC;
#pragma unroll
        for (int r = 0; r < RANK; ++r) {
            float f = Fxy[r * RES * RES + iy * RES + ix]
                    + Fxz[r * RES * RES + iz * RES + ix]
                    + Fyz[r * RES * RES + iz * RES + iy];
            row[r] = (_Float16)f;
        }
        row[8]  = (_Float16)x0;
        row[9]  = (_Float16)x1;
        row[10] = (_Float16)x2;
        float xv[3] = {x0, x1, x2};
#pragma unroll
        for (int l = 0; l < PE_L; ++l) {
            float fr = (float)(1 << l);
#pragma unroll
            for (int d = 0; d < 3; ++d) {
                float a = xv[d] * fr;
                row[11 + l * 6 + d]     = (_Float16)__sinf(a);
                row[11 + l * 6 + 3 + d] = (_Float16)__cosf(a);
            }
        }
        for (int c = 8 + 3 + 6 * PE_L; c < 64; ++c) row[c] = (_Float16)0.0f;
    }

    // ---- layer 0: 64 -> 256 ----
    wait_async_all();
    __syncthreads();
    mlp_layer<2>(hb, wb, B0, lane);

    // ---- hidden layers 1..3: 256 -> 256 ----
    for (int L = 0; L < 3; ++L) {
        __syncthreads();   // all waves done reading wb
        stage_async(wb, Wpk + OFF_L1 + L * SZ_LH, SZ_LH, tid);
        wait_async_all();
        __syncthreads();
        const float* Bl = (L == 0) ? B1 : (L == 1) ? B2 : B3;
        mlp_layer<8>(hb, wb, Bl, lane);
    }

    // ---- heads: sigma (softplus) + rgb (sigmoid), one n-tile ----
    __syncthreads();
    stage_async(wb, Wpk + OFF_HD, SZ_HEAD, tid);
    wait_async_all();
    __syncthreads();
    {
        const int ln15 = lane & 15;
        const int klo  = (lane < 16) ? 0 : 8;
        float b = (ln15 == 0) ? Bsig[0] : (ln15 <= 3 ? Brgb[ln15 - 1] : 0.0f);
        v8f c;
#pragma unroll
        for (int r = 0; r < 8; ++r) c[r] = b;
#pragma unroll
        for (int kt = 0; kt < 8; ++kt) {
            const v4u* p = (const v4u*)(hb + ln15 * ENC + kt * 32 + klo);
            Frag fa; fa.u[0] = p[0]; fa.u[1] = p[2];
            const v4u* q = (const v4u*)(wb + (kt * 32 + lane) * 16);
            Frag fb; fb.u[0] = q[0]; fb.u[1] = q[1];
            c = __builtin_amdgcn_wmma_f32_16x16x32_f16(
                    false, fa.v, false, fb.v, (short)0, c, false, false);
        }
#pragma unroll
        for (int r = 0; r < 8; ++r) {
            int mm  = r + ((lane < 16) ? 0 : 8);
            int idx = tileBase + mm;
            float v = c[r];
            if (ln15 == 0) {
                float sp = (v > 20.0f) ? v : log1pf(__expf(v));
                Out[3 * nPoints + idx] = sp;              // sigma
            } else if (ln15 <= 3) {
                Out[idx * 3 + (ln15 - 1)] = 1.0f / (1.0f + __expf(-v)); // rgb
            }
        }
    }
}

// ---------------------------------------------------------------------------
extern "C" void kernel_launch(void* const* d_in, const int* in_sizes, int n_in,
                              void* d_out, int out_size, void* d_ws, size_t ws_size,
                              hipStream_t stream)
{
    const float* X    = (const float*)d_in[0];
    const float* Fxy  = (const float*)d_in[1];
    const float* Fxz  = (const float*)d_in[2];
    const float* Fyz  = (const float*)d_in[3];
    const float* W0   = (const float*)d_in[4];
    const float* b0   = (const float*)d_in[5];
    const float* W1   = (const float*)d_in[6];
    const float* b1   = (const float*)d_in[7];
    const float* W2   = (const float*)d_in[8];
    const float* b2   = (const float*)d_in[9];
    const float* W3   = (const float*)d_in[10];
    const float* b3   = (const float*)d_in[11];
    const float* Wsig = (const float*)d_in[12];
    const float* bsig = (const float*)d_in[13];
    const float* Wrgb = (const float*)d_in[14];
    const float* brgb = (const float*)d_in[15];

    _Float16* wpk = (_Float16*)d_ws;   // ~425 KB of packed f16 weights
    const int nPoints = in_sizes[0] / 3;

    // pack weights into WMMA B-fragment order (f16)
    pack_w<<<(SZ_L0 + 255) / 256, 256, 0, stream>>>(W0, wpk + OFF_L0, 47, 2, ENC);
    pack_w<<<(SZ_LH + 255) / 256, 256, 0, stream>>>(W1, wpk + OFF_L1, ENC, 8, ENC);
    pack_w<<<(SZ_LH + 255) / 256, 256, 0, stream>>>(W2, wpk + OFF_L2, ENC, 8, ENC);
    pack_w<<<(SZ_LH + 255) / 256, 256, 0, stream>>>(W3, wpk + OFF_L3, ENC, 8, ENC);
    pack_head<<<(SZ_HEAD + 255) / 256, 256, 0, stream>>>(Wsig, Wrgb, wpk + OFF_HD);

    // fused MLP: 8 waves/block, 128 points/block
    const size_t smemBytes = (size_t)(8 * 16 * ENC * 2) + (size_t)(SZ_LH * 2); // 64KB + 128KB
    hipFuncSetAttribute((const void*)tensorf_mlp,
                        hipFuncAttributeMaxDynamicSharedMemorySize,
                        (int)smemBytes);
    tensorf_mlp<<<nPoints / 128, 256, smemBytes, stream>>>(
        X, Fxy, Fxz, Fyz, wpk, b0, b1, b2, b3, bsig, brgb,
        (float*)d_out, nPoints);
}
